// SetAbstractionBlock_33079838113812
// MI455X (gfx1250) — matrix operand order repaired and checked by reading, not compiled, not run on hardware
//
#include <hip/hip_runtime.h>
#include <hip/hip_bf16.h>

// ---------------------------------------------------------------------------
// PointNet++ SetAbstraction block for gfx1250 (MI455X), wave32 + WMMA.
//   K1: FPS          (1 workgroup / batch, point cloud resident in LDS)
//   K2: ball query   (LDS-staged chunks, ascending scan == sort semantics)
//   K0: weight prep  (fold BN into f16 weights + f32 bias)
//   K3: gather + 3-layer MLP via v_wmma_f32_16x16x32_f16 + fused max-pool
//       (wave-uniform scalar job loops, B-fragments hoisted per wave)
// ---------------------------------------------------------------------------

typedef __attribute__((ext_vector_type(8)))  _Float16 v8h;
typedef __attribute__((ext_vector_type(16))) _Float16 v16h;
typedef __attribute__((ext_vector_type(8)))  float    v8f;

union AB16 { v16h v; v8h h[2]; };

#define NPOINT  2048
#define NSAMPLE 32
#define NPTS    8192
#define NBATCH  8
#define CFEAT   64
#define RADIUS  0.2f

// -------------------------- K1: farthest point sampling --------------------
__global__ __launch_bounds__(1024) void fps_kernel(const float* __restrict__ xyz,
                                                   float* __restrict__ new_xyz)
{
    __shared__ float sx[NPTS], sy[NPTS], sz[NPTS];
    __shared__ float dist[NPTS];
    __shared__ float rv[32];
    __shared__ int   ri[32];
    __shared__ int   cur;

    const int b    = blockIdx.x;
    const int tid  = threadIdx.x;
    const int lane = tid & 31;
    const int warp = tid >> 5;
    const float* base = xyz + (size_t)b * NPTS * 3;

    for (int i = tid; i < NPTS; i += 1024) {
        sx[i] = base[i * 3 + 0];
        sy[i] = base[i * 3 + 1];
        sz[i] = base[i * 3 + 2];
        dist[i] = 1e10f;
    }
    if (tid == 0) cur = 0;
    __syncthreads();

    for (int t = 0; t < NPOINT; ++t) {
        const int f = cur;
        const float cx = sx[f], cy = sy[f], cz = sz[f];
        if (tid == 0) {
            float* o = new_xyz + ((size_t)b * NPOINT + t) * 3;
            o[0] = cx; o[1] = cy; o[2] = cz;
        }
        float bv = -1.0f;
        int   bi = 0x7fffffff;
#pragma unroll
        for (int u = 0; u < NPTS / 1024; ++u) {
            const int i = tid + u * 1024;
            const float dx = sx[i] - cx, dy = sy[i] - cy, dz = sz[i] - cz;
            const float d  = dx * dx + dy * dy + dz * dz;
            const float nd = fminf(dist[i], d);
            dist[i] = nd;
            if (nd > bv || (nd == bv && i < bi)) { bv = nd; bi = i; }
        }
        // wave32 reduce (argmax, lowest index on tie)
#pragma unroll
        for (int off = 16; off > 0; off >>= 1) {
            const float ov = __shfl_down(bv, off, 32);
            const int   oi = __shfl_down(bi, off, 32);
            if (ov > bv || (ov == bv && oi < bi)) { bv = ov; bi = oi; }
        }
        if (lane == 0) { rv[warp] = bv; ri[warp] = bi; }
        __syncthreads();
        if (warp == 0) {
            bv = rv[lane]; bi = ri[lane];
#pragma unroll
            for (int off = 16; off > 0; off >>= 1) {
                const float ov = __shfl_down(bv, off, 32);
                const int   oi = __shfl_down(bi, off, 32);
                if (ov > bv || (ov == bv && oi < bi)) { bv = ov; bi = oi; }
            }
            if (lane == 0) cur = bi;
        }
        __syncthreads();
    }
}

// -------------------------- K2: ball query ----------------------------------
__global__ __launch_bounds__(256) void ballq_kernel(const float* __restrict__ xyz,
                                                    const float* __restrict__ new_xyz,
                                                    int* __restrict__ ball_idx)
{
    __shared__ float px[2048], py[2048], pz[2048];

    const int b      = blockIdx.y;
    const int tid    = threadIdx.x;
    const int center = blockIdx.x * 256 + tid;

    const float cx = new_xyz[((size_t)b * NPOINT + center) * 3 + 0];
    const float cy = new_xyz[((size_t)b * NPOINT + center) * 3 + 1];
    const float cz = new_xyz[((size_t)b * NPOINT + center) * 3 + 2];

    int* out = ball_idx + ((size_t)b * NPOINT + center) * NSAMPLE;
    int cnt = 0, first = 0;
    const float r2 = RADIUS * RADIUS;
    const float* base = xyz + (size_t)b * NPTS * 3;

    for (int ch = 0; ch < NPTS / 2048; ++ch) {
        for (int j = tid; j < 2048; j += 256) {
            const int p = ch * 2048 + j;
            px[j] = base[p * 3 + 0];
            py[j] = base[p * 3 + 1];
            pz[j] = base[p * 3 + 2];
        }
        __syncthreads();
        if (cnt < NSAMPLE) {
            for (int j = 0; j < 2048; ++j) {
                const float dx = px[j] - cx, dy = py[j] - cy, dz = pz[j] - cz;
                const float d2 = dx * dx + dy * dy + dz * dz;
                if (d2 <= r2) {
                    const int p = ch * 2048 + j;
                    if (cnt == 0) first = p;
                    out[cnt++] = p;
                    if (cnt == NSAMPLE) break;
                }
            }
        }
        __syncthreads();
    }
    for (int k = cnt; k < NSAMPLE; ++k) out[k] = first;   // pad with first hit
}

// -------------------------- K0: fold BN into f16 weights --------------------
__global__ void prep_kernel(const float* __restrict__ w1, const float* __restrict__ b1,
                            const float* __restrict__ g1, const float* __restrict__ bt1,
                            const float* __restrict__ w2, const float* __restrict__ b2,
                            const float* __restrict__ g2, const float* __restrict__ bt2,
                            const float* __restrict__ w3, const float* __restrict__ b3,
                            const float* __restrict__ g3, const float* __restrict__ bt3,
                            _Float16* __restrict__ W1p, float* __restrict__ B1p,
                            _Float16* __restrict__ W2p, float* __restrict__ B2p,
                            _Float16* __restrict__ W3p, float* __restrict__ B3p)
{
    const float s   = rsqrtf(1.0f + 1e-5f);
    const int   tid = blockIdx.x * blockDim.x + threadIdx.x;
    const int   nth = gridDim.x * blockDim.x;

    for (int e = tid; e < 64 * 96; e += nth) {               // 67 -> pad 96
        const int o = e / 96, c = e % 96;
        const float v = (c < 67) ? w1[o * 67 + c] * (g1[o] * s) : 0.0f;
        W1p[e] = (_Float16)v;
    }
    for (int e = tid; e < 64 * 64; e += nth) {
        const int o = e >> 6;
        W2p[e] = (_Float16)(w2[e] * (g2[o] * s));
    }
    for (int e = tid; e < 128 * 64; e += nth) {
        const int o = e >> 6;
        W3p[e] = (_Float16)(w3[e] * (g3[o] * s));
    }
    for (int o = tid; o < 64;  o += nth) B1p[o] = b1[o] * (g1[o] * s) + bt1[o];
    for (int o = tid; o < 64;  o += nth) B2p[o] = b2[o] * (g2[o] * s) + bt2[o];
    for (int o = tid; o < 128; o += nth) B3p[o] = b3[o] * (g3[o] * s) + bt3[o];
}

// -------------------------- WMMA fragment helpers ---------------------------
// A-frag (16x32 f16, ISA 7.12.2): lane l<16 holds row M=l, K {k0..k0+7} then
// {k0+16..k0+23}; lane l>=16 holds row M=l-16, K shifted by +8.
__device__ __forceinline__ v16h ldsA(const _Float16* X, int ldx, int row0, int k0, int lane)
{
    const int l = lane & 15, hi = (lane >> 4) & 1;
    const _Float16* p = X + (row0 + l) * ldx + k0 + hi * 8;
    AB16 u;
    u.h[0] = *(const v8h*)(p);
    u.h[1] = *(const v8h*)(p + 16);
    return u.v;
}
// B-frag (32x16 f16): lane l<16 = column N=l, K {k0..k0+7} then {k0+16..k0+23}; lane>=16 K+8.
// Weights are row-major W'[n][k], so each lane reads two 16-byte runs starting 32 bytes apart.
__device__ __forceinline__ v16h gldB(const _Float16* W, int ldw, int n0, int k0, int lane)
{
    const int l = lane & 15, hi = (lane >> 4) & 1;
    const _Float16* p = W + (n0 + l) * ldw + k0 + hi * 8;
    AB16 u;
    u.h[0] = *(const v8h*)(p);
    u.h[1] = *(const v8h*)(p + 16);
    return u.v;
}
__device__ __forceinline__ v8f wmma16x16x32(v16h a, v16h b, v8f c)
{
    return __builtin_amdgcn_wmma_f32_16x16x32_f16(false, a, false, b, (short)0, c, false, false);
}

// -------------------------- K3: gather + MLP + maxpool ----------------------
#define TC    8              // centers per workgroup
#define ROWS  256            // TC * NSAMPLE
#define LDX0  104            // 96 cols + 8 pad (keeps rows 16B aligned)
#define LDX1  72             // 64 cols + 8 pad

__global__ __launch_bounds__(256) void mlp_kernel(
    const float* __restrict__ xyz, const float* __restrict__ feats,
    const float* __restrict__ new_xyz, const int* __restrict__ ball_idx,
    const _Float16* __restrict__ W1, const float* __restrict__ B1,
    const _Float16* __restrict__ W2, const float* __restrict__ B2,
    const _Float16* __restrict__ W3, const float* __restrict__ B3,
    float* __restrict__ out_feats)
{
    __shared__ _Float16     X0[ROWS * LDX0];   // 53248 B  layer-1 input (67 -> pad 96)
    __shared__ _Float16     X1[ROWS * LDX1];   // 36864 B
    __shared__ _Float16     X2[ROWS * LDX1];   // 36864 B
    __shared__ unsigned int pool[TC * 128];    //  4096 B  maxpool (relu >= 0 -> uint order)

    const int b    = blockIdx.y;
    const int c0   = blockIdx.x * TC;
    const int tid  = threadIdx.x;
    const int lane = tid & 31;
    // SGPR wave id: keeps the per-wave job loops scalar (SALU control flow,
    // EXEC stays all-ones through every v_wmma as required by ISA 7.12).
    const int wave = __builtin_amdgcn_readfirstlane(tid >> 5);

    for (int i = tid; i < TC * 128; i += 256) pool[i] = 0u;

    // ---- gather: one row (center-sample) per thread -------------------------
    {
        const int r      = tid;
        const int cl     = r >> 5;
        const int k      = r & 31;
        const int center = c0 + cl;
        const int j      = ball_idx[((size_t)b * NPOINT + center) * NSAMPLE + k];
        const float inv_r = 1.0f / RADIUS;
        const float nx = new_xyz[((size_t)b * NPOINT + center) * 3 + 0];
        const float ny = new_xyz[((size_t)b * NPOINT + center) * 3 + 1];
        const float nz = new_xyz[((size_t)b * NPOINT + center) * 3 + 2];
        const float* pj = xyz + ((size_t)b * NPTS + j) * 3;
        _Float16* row = X0 + r * LDX0;
        row[0] = (_Float16)((pj[0] - nx) * inv_r);
        row[1] = (_Float16)((pj[1] - ny) * inv_r);
        row[2] = (_Float16)((pj[2] - nz) * inv_r);
        const float* fr = feats + ((size_t)b * NPTS + j) * CFEAT;
#pragma unroll
        for (int c = 0; c < CFEAT; c += 4) {
            const float4 v = *(const float4*)(fr + c);
            row[3 + c + 0] = (_Float16)v.x;
            row[3 + c + 1] = (_Float16)v.y;
            row[3 + c + 2] = (_Float16)v.z;
            row[3 + c + 3] = (_Float16)v.w;
        }
#pragma unroll
        for (int c = 67; c < 96; ++c) row[c] = (_Float16)0.0f;
    }
    __syncthreads();

    // ---- layer 1: 67(->96) -> 64. ct is wave-invariant (8 waves, 4 col tiles):
    //      waves {w, w+4} share ct = w&3 and split the 16 row tiles. ---------
    {
        const int ct = wave & 3;
        const int n  = ct * 16 + (lane & 15);
        const v16h wb0 = gldB(W1, 96, ct * 16,  0, lane);
        const v16h wb1 = gldB(W1, 96, ct * 16, 32, lane);
        const v16h wb2 = gldB(W1, 96, ct * 16, 64, lane);
        const float bv = B1[n];
        const int   hb = (lane >> 4) << 3;
        for (int rt = (wave >> 2); rt < 16; rt += 2) {
            v8f acc;
#pragma unroll
            for (int i = 0; i < 8; ++i) acc[i] = bv;
            acc = wmma16x16x32(ldsA(X0, LDX0, rt * 16,  0, lane), wb0, acc);
            acc = wmma16x16x32(ldsA(X0, LDX0, rt * 16, 32, lane), wb1, acc);
            acc = wmma16x16x32(ldsA(X0, LDX0, rt * 16, 64, lane), wb2, acc);
            const int rbase = rt * 16 + hb;
#pragma unroll
            for (int p = 0; p < 8; ++p)
                X1[(rbase + p) * LDX1 + n] = (_Float16)fmaxf(acc[p], 0.0f);
        }
    }
    __syncthreads();

    // ---- layer 2: 64 -> 64 --------------------------------------------------
    {
        const int ct = wave & 3;
        const int n  = ct * 16 + (lane & 15);
        const v16h wb0 = gldB(W2, 64, ct * 16,  0, lane);
        const v16h wb1 = gldB(W2, 64, ct * 16, 32, lane);
        const float bv = B2[n];
        const int   hb = (lane >> 4) << 3;
        for (int rt = (wave >> 2); rt < 16; rt += 2) {
            v8f acc;
#pragma unroll
            for (int i = 0; i < 8; ++i) acc[i] = bv;
            acc = wmma16x16x32(ldsA(X1, LDX1, rt * 16,  0, lane), wb0, acc);
            acc = wmma16x16x32(ldsA(X1, LDX1, rt * 16, 32, lane), wb1, acc);
            const int rbase = rt * 16 + hb;
#pragma unroll
            for (int p = 0; p < 8; ++p)
                X2[(rbase + p) * LDX1 + n] = (_Float16)fmaxf(acc[p], 0.0f);
        }
    }
    __syncthreads();

    // ---- layer 3: 64 -> 128, fused ReLU + maxpool over 32 samples.
    //      ct = wave (8 col tiles); each wave walks all 16 row tiles. --------
    {
        const int ct = wave;
        const int n  = ct * 16 + (lane & 15);
        const v16h wb0 = gldB(W3, 64, ct * 16,  0, lane);
        const v16h wb1 = gldB(W3, 64, ct * 16, 32, lane);
        const float bv = B3[n];
        for (int rt = 0; rt < 16; ++rt) {
            v8f acc;
#pragma unroll
            for (int i = 0; i < 8; ++i) acc[i] = bv;
            acc = wmma16x16x32(ldsA(X2, LDX1, rt * 16,  0, lane), wb0, acc);
            acc = wmma16x16x32(ldsA(X2, LDX1, rt * 16, 32, lane), wb1, acc);
            float m = 0.0f;                  // ReLU floor doubles as pool init
#pragma unroll
            for (int p = 0; p < 8; ++p) m = fmaxf(m, acc[p]);
            const int cl = rt >> 1;          // 2 row tiles per center
            atomicMax(&pool[cl * 128 + n], __float_as_uint(m));
        }
    }
    __syncthreads();

    for (int e = tid; e < TC * 128; e += 256) {
        const int cl = e >> 7, n = e & 127;
        out_feats[(((size_t)b * NPOINT + c0 + cl) << 7) + n] = __uint_as_float(pool[e]);
    }
}

// -------------------------- host launcher -----------------------------------
extern "C" void kernel_launch(void* const* d_in, const int* in_sizes, int n_in,
                              void* d_out, int out_size, void* d_ws, size_t ws_size,
                              hipStream_t stream)
{
    const float* xyz   = (const float*)d_in[0];
    const float* feats = (const float*)d_in[1];
    const float* w1 = (const float*)d_in[2],  *b1 = (const float*)d_in[3];
    const float* g1 = (const float*)d_in[4],  *bt1 = (const float*)d_in[5];
    const float* w2 = (const float*)d_in[6],  *b2 = (const float*)d_in[7];
    const float* g2 = (const float*)d_in[8],  *bt2 = (const float*)d_in[9];
    const float* w3 = (const float*)d_in[10], *b3 = (const float*)d_in[11];
    const float* g3 = (const float*)d_in[12], *bt3 = (const float*)d_in[13];

    float* out       = (float*)d_out;
    float* new_xyz   = out;                                  // B*2048*3
    float* new_feats = out + (size_t)NBATCH * NPOINT * 3;    // B*2048*128

    char* ws = (char*)d_ws;
    size_t off = 0;
    int*      ball = (int*)(ws + off);       off += (size_t)NBATCH * NPOINT * NSAMPLE * 4; // 2 MB
    _Float16* W1p  = (_Float16*)(ws + off);  off += 64 * 96 * 2;
    _Float16* W2p  = (_Float16*)(ws + off);  off += 64 * 64 * 2;
    _Float16* W3p  = (_Float16*)(ws + off);  off += 128 * 64 * 2;
    float*    B1p  = (float*)(ws + off);     off += 256;
    float*    B2p  = (float*)(ws + off);     off += 256;
    float*    B3p  = (float*)(ws + off);     off += 512;

    prep_kernel<<<32, 256, 0, stream>>>(w1, b1, g1, bt1, w2, b2, g2, bt2,
                                        w3, b3, g3, bt3,
                                        W1p, B1p, W2p, B2p, W3p, B3p);
    fps_kernel<<<NBATCH, 1024, 0, stream>>>(xyz, new_xyz);
    ballq_kernel<<<dim3(NPOINT / 256, NBATCH), 256, 0, stream>>>(xyz, new_xyz, ball);
    mlp_kernel<<<dim3(NPOINT / TC, NBATCH), 256, 0, stream>>>(
        xyz, feats, new_xyz, ball, W1p, B1p, W2p, B2p, W3p, B3p, new_feats);
}
